// SelfAttention_23235773071931
// MI455X (gfx1250) — compile-verified, matrix operand order checked by reading, not run on hardware
//
#include <hip/hip_runtime.h>

#define D_DIM 1024
#define S_LEN 2048
#define B_N   4
#define M_ROWS (B_N * S_LEN)   // 8192 total rows across batches
#define QT    32               // query rows per attention block

typedef __attribute__((ext_vector_type(16))) _Float16 v16h;
typedef __attribute__((ext_vector_type(8)))  _Float16 v8h;
typedef __attribute__((ext_vector_type(8)))  float    v8f;

static __device__ __forceinline__ v8f wmma16x16x32(v16h a, v16h b, v8f c) {
  // 8 args: (neg_a, A, neg_b, B, c_mod, C, reuse_a, reuse_b)
  return __builtin_amdgcn_wmma_f32_16x16x32_f16(false, a, false, b, (short)0, c,
                                                false, false);
}

// Load a fragment from a precomputed per-lane base pointer. The pointer already
// encodes (row/col = lane&15)*ld + kb where kb = (lane>=16)?8:0. Tile and
// k-step deltas are applied by the caller as constant offsets so they fold
// into the instruction's immediate offset field.
static __device__ __forceinline__ v16h load_frag_at(const _Float16* p) {
  v8h lo = *(const v8h*)p;          // k = kb .. kb+7
  v8h hi = *(const v8h*)(p + 16);   // k = kb+16 .. kb+23
  v16h r;
#pragma unroll
  for (int i = 0; i < 8; ++i) { r[i] = lo[i]; r[i + 8] = hi[i]; }
  return r;
}

__global__ void cvt_f32_f16(const float* __restrict__ in,
                            _Float16* __restrict__ out, int n) {
  int i = blockIdx.x * blockDim.x + threadIdx.x;
  if (i < n) out[i] = (_Float16)in[i];
}

// Y = x @ W^T + bias, 2(M) x 4(N) register blocking per wave (64 acc VGPRs,
// transient B fragments): 6 fragment loads -> 8 WMMAs per k-step.
// Block = 8 waves arranged 4(M) x 2(N) -> 128x128 output tile.
// vmode==0: Y row-major f16 (Q/K). vmode==1: Y transposed [b][e][s] f16 (V).
__global__ void proj_gemm(const _Float16* __restrict__ xh,
                          const _Float16* __restrict__ Wh,
                          const float* __restrict__ bias,
                          _Float16* __restrict__ out, int vmode) {
  const int lane = threadIdx.x & 31;
  const int wave = threadIdx.x >> 5;
  const int row0 = blockIdx.y * 128 + (wave & 3) * 32;   // 2 M-tiles of 16
  const int n0   = blockIdx.x * 128 + (wave >> 2) * 64;  // 4 N-tiles of 16
  const int col = lane & 15;
  const int kb = (lane >> 4) << 3;

  const _Float16* pA = xh + (size_t)(row0 + col) * D_DIM + kb;
  const _Float16* pB = Wh + (size_t)(n0 + col) * D_DIM + kb;

  v8f acc[2][4] = {};
  for (int k0 = 0; k0 < D_DIM; k0 += 32, pA += 32, pB += 32) {
    v16h a0 = load_frag_at(pA);
    v16h a1 = load_frag_at(pA + 16 * D_DIM);
#pragma unroll
    for (int j = 0; j < 4; ++j) {
      v16h b = load_frag_at(pB + j * 16 * D_DIM);   // transient B fragment
      acc[0][j] = wmma16x16x32(a0, b, acc[0][j]);
      acc[1][j] = wmma16x16x32(a1, b, acc[1][j]);
    }
  }

  const int mb = (lane >> 4) << 3;
#pragma unroll
  for (int j = 0; j < 4; ++j) {
    const int ncol = n0 + 16 * j + col;
    const float bv = bias[ncol];
#pragma unroll
    for (int i = 0; i < 2; ++i) {
      const int mbase = row0 + 16 * i + mb;
      if (vmode == 0) {
#pragma unroll
        for (int r = 0; r < 8; ++r)
          out[(size_t)(mbase + r) * D_DIM + ncol] = (_Float16)(acc[i][j][r] + bv);
      } else {
        const int bb = mbase >> 11;               // batch (tile-uniform)
        const int s = mbase & (S_LEN - 1);
        v8h pk;
#pragma unroll
        for (int r = 0; r < 8; ++r) pk[r] = (_Float16)(acc[i][j][r] + bv);
        *(v8h*)(out + ((size_t)bb * D_DIM + ncol) * S_LEN + s) = pk;
      }
    }
  }
}

// One block = one (batch, 32-query stripe). 8 waves, 256 threads.
// LDS: sc[32][2048] f32 scores (256 KB, CDNA5-only capacity) + red[256] f32.
__global__ void attn_kernel(const _Float16* __restrict__ Qh,
                            const _Float16* __restrict__ Kh,
                            const _Float16* __restrict__ Vth,
                            float* __restrict__ out) {
  extern __shared__ char smem[];
  float* sc  = (float*)smem;                                        // QT*S_LEN f32
  float* red = (float*)(smem + (size_t)QT * S_LEN * sizeof(float)); // 256 f32

  const int tid = threadIdx.x;
  const int lane = tid & 31;
  const int wave = tid >> 5;
  const int q0 = blockIdx.x * QT;
  const int bb = blockIdx.y;
  const int grow0 = bb * S_LEN + q0;   // global row of this Q stripe
  const int col = lane & 15;
  const int kb = (lane >> 4) << 3;
  const int mb = (lane >> 4) << 3;

  // ---- Phase 1: scores = scale*Q K^T -> LDS. Two passes of 1024 keys;
  // per pass each wave owns 2(M) x 8(keys) tiles, transient B fragments:
  // 10 fragment loads -> 16 WMMAs per k-step.
  const float scale = 0.03125f;        // 1/sqrt(1024)
  for (int pass = 0; pass < 2; ++pass) {
    const _Float16* pQ = Qh + (size_t)(grow0 + col) * D_DIM + kb;
    const _Float16* pK = Kh +
        (size_t)(bb * S_LEN + pass * 1024 + wave * 128 + col) * D_DIM + kb;
    v8f acc[2][8] = {};
    for (int k0 = 0; k0 < D_DIM; k0 += 32, pQ += 32, pK += 32) {
      v16h a0 = load_frag_at(pQ);
      v16h a1 = load_frag_at(pQ + 16 * D_DIM);
#pragma unroll
      for (int t = 0; t < 8; ++t) {
        v16h b = load_frag_at(pK + t * 16 * D_DIM);   // transient B fragment
        acc[0][t] = wmma16x16x32(a0, b, acc[0][t]);
        acc[1][t] = wmma16x16x32(a1, b, acc[1][t]);
      }
    }
#pragma unroll
    for (int m = 0; m < 2; ++m)
#pragma unroll
      for (int t = 0; t < 8; ++t) {
        const int key = pass * 1024 + (wave * 8 + t) * 16 + col;
#pragma unroll
        for (int r = 0; r < 8; ++r)
          sc[(16 * m + mb + r) * S_LEN + key] = acc[m][t][r] * scale;
      }
  }
  __syncthreads();

  // ---- Phase 2: softmax over each of QT rows (8 threads per row) ----
  {
    const int row = tid >> 3;
    const int sub = tid & 7;
    float* srow = sc + row * S_LEN;
    float mx = -1e30f;
    for (int j = sub; j < S_LEN; j += 8) mx = fmaxf(mx, srow[j]);
    red[tid] = mx;
    __syncthreads();
    if (sub == 0) {
      float m2 = red[tid];
      for (int i = 1; i < 8; ++i) m2 = fmaxf(m2, red[tid + i]);
      red[tid] = m2;
    }
    __syncthreads();
    const float rmax = red[row << 3];
    __syncthreads();
    float sum = 0.f;
    for (int j = sub; j < S_LEN; j += 8) {
      float p = __expf(srow[j] - rmax);
      srow[j] = p;
      sum += p;
    }
    red[tid] = sum;
    __syncthreads();
    if (sub == 0) {
      float s2 = 0.f;
      for (int i = 0; i < 8; ++i) s2 += red[tid + i];
      red[tid] = s2;
    }
    __syncthreads();
    const float inv = 1.0f / red[row << 3];
    __syncthreads();
    for (int j = sub; j < S_LEN; j += 8) srow[j] *= inv;
  }
  __syncthreads();

  // ---- Phase 3: out = P @ V (Vth: [b][e][s], k-major for B operand).
  // Each wave owns 2(M) x 8(e) tiles over a 128-wide e-slice; B transient.
  {
    const int e0w = wave * 128;
    const _Float16* pV = Vth + (size_t)bb * D_DIM * S_LEN +
                         (size_t)(e0w + col) * S_LEN + kb;
    const float* pr0 = sc + (size_t)(lane & 15) * S_LEN + kb;
    v8f acc[2][8] = {};
    for (int k0 = 0; k0 < S_LEN; k0 += 32, pV += 32, pr0 += 32) {
      v16h a[2];
#pragma unroll
      for (int m = 0; m < 2; ++m) {
        const float* pr = pr0 + 16 * m * S_LEN;
#pragma unroll
        for (int i = 0; i < 8; ++i) {
          a[m][i]     = (_Float16)pr[i];        // keys kb..kb+7
          a[m][i + 8] = (_Float16)pr[i + 16];   // keys kb+16..kb+23
        }
      }
#pragma unroll
      for (int t = 0; t < 8; ++t) {
        v16h b = load_frag_at(pV + t * 16 * S_LEN);   // transient B fragment
        acc[0][t] = wmma16x16x32(a[0], b, acc[0][t]);
        acc[1][t] = wmma16x16x32(a[1], b, acc[1][t]);
      }
    }
    const size_t obase = ((size_t)bb * S_LEN + q0) * D_DIM;
#pragma unroll
    for (int m = 0; m < 2; ++m)
#pragma unroll
      for (int t = 0; t < 8; ++t) {
        const int e = e0w + 16 * t + col;
#pragma unroll
        for (int r = 0; r < 8; ++r)
          out[obase + (size_t)(16 * m + mb + r) * D_DIM + e] = acc[m][t][r];
      }
  }
}

extern "C" void kernel_launch(void* const* d_in, const int* in_sizes, int n_in,
                              void* d_out, int out_size, void* d_ws, size_t ws_size,
                              hipStream_t stream) {
  (void)in_sizes; (void)n_in; (void)out_size; (void)ws_size;
  const float* x  = (const float*)d_in[0];
  const float* Wq = (const float*)d_in[1];
  const float* bq = (const float*)d_in[2];
  const float* Wk = (const float*)d_in[3];
  const float* bk = (const float*)d_in[4];
  const float* Wv = (const float*)d_in[5];
  const float* bv = (const float*)d_in[6];
  float* out = (float*)d_out;

  // Workspace carve-up (f16): x, Wq, Wk, Wv, Q, K, V^T  (~70 MB total)
  _Float16* xh  = (_Float16*)d_ws;
  _Float16* Wqh = xh  + (size_t)M_ROWS * D_DIM;
  _Float16* Wkh = Wqh + (size_t)D_DIM * D_DIM;
  _Float16* Wvh = Wkh + (size_t)D_DIM * D_DIM;
  _Float16* Qh  = Wvh + (size_t)D_DIM * D_DIM;
  _Float16* Kh  = Qh  + (size_t)M_ROWS * D_DIM;
  _Float16* Vth = Kh  + (size_t)M_ROWS * D_DIM;

  const int nx = M_ROWS * D_DIM;
  cvt_f32_f16<<<(nx + 255) / 256, 256, 0, stream>>>(x, xh, nx);
  const int nw = D_DIM * D_DIM;
  cvt_f32_f16<<<(nw + 255) / 256, 256, 0, stream>>>(Wq, Wqh, nw);
  cvt_f32_f16<<<(nw + 255) / 256, 256, 0, stream>>>(Wk, Wkh, nw);
  cvt_f32_f16<<<(nw + 255) / 256, 256, 0, stream>>>(Wv, Wvh, nw);

  proj_gemm<<<dim3(D_DIM / 128, M_ROWS / 128), 256, 0, stream>>>(xh, Wqh, bq, Qh, 0);
  proj_gemm<<<dim3(D_DIM / 128, M_ROWS / 128), 256, 0, stream>>>(xh, Wkh, bk, Kh, 0);
  proj_gemm<<<dim3(D_DIM / 128, M_ROWS / 128), 256, 0, stream>>>(xh, Wvh, bv, Vth, 1);

  const size_t smem = (size_t)QT * S_LEN * sizeof(float) + 256 * sizeof(float);
  attn_kernel<<<dim3(S_LEN / QT, B_N), 256, smem, stream>>>(Qh, Kh, Vth, out);
}
